// CausalSelfAttention_84464826843572
// MI455X (gfx1250) — compile-verified, matrix operand order checked by reading, not run on hardware
//
#include <hip/hip_runtime.h>
#include <hip/hip_bf16.h>
#include <math.h>

typedef __attribute__((ext_vector_type(16))) __bf16 bf16x16;
typedef __attribute__((ext_vector_type(8)))  __bf16 bf16x8;
typedef __attribute__((ext_vector_type(4)))  __bf16 bf16x4;
typedef __attribute__((ext_vector_type(2)))  __bf16 bf16x2;
typedef __attribute__((ext_vector_type(8)))  float  f32x8;
typedef __attribute__((ext_vector_type(4)))  float  f32x4;

// D = A(16x32 bf16) x B(32x16 bf16) + C(16x16 f32)
__device__ __forceinline__ f32x8 wmma_bf16(bf16x16 a, bf16x16 b, f32x8 c) {
    return __builtin_amdgcn_wmma_f32_16x16x32_bf16(false, a, false, b, (short)0, c, false, false);
}

// Fragment load from LDS (bf16, per-lane: 8 contiguous @ kbase, 8 contiguous @ kbase+16)
__device__ __forceinline__ bf16x16 frag_lds(const __bf16* base, int kbase) {
    bf16x8 lo = *(const bf16x8*)(base + kbase);
    bf16x8 hi = *(const bf16x8*)(base + kbase + 16);
    return __builtin_shufflevector(lo, hi, 0,1,2,3,4,5,6,7,8,9,10,11,12,13,14,15);
}

// Fragment straight from a global f32 row (contiguous contraction dim), packed cvt to bf16
__device__ __forceinline__ bf16x16 frag_gbl_f32(const float* base, int kbase) {
    f32x4 a0 = *(const f32x4*)(base + kbase);
    f32x4 a1 = *(const f32x4*)(base + kbase + 4);
    f32x4 b0 = *(const f32x4*)(base + kbase + 16);
    f32x4 b1 = *(const f32x4*)(base + kbase + 20);
    f32x8 lo = __builtin_shufflevector(a0, a1, 0,1,2,3,4,5,6,7);
    f32x8 hi = __builtin_shufflevector(b0, b1, 0,1,2,3,4,5,6,7);
    bf16x8 l = __builtin_convertvector(lo, bf16x8);
    bf16x8 h = __builtin_convertvector(hi, bf16x8);
    return __builtin_shufflevector(l, h, 0,1,2,3,4,5,6,7,8,9,10,11,12,13,14,15);
}

// ---------------------------------------------------------------------------
// Kernel 1: qkv = x @ W    (M=8192, K=1024, N=3072), fp32 in/out, bf16 WMMA.
// Block tile 128x128, 8 waves of 32x64, double-buffered LDS staging.
// ---------------------------------------------------------------------------
#define LDA 40   // bf16 elements per LDS row (32 + 8 pad) -> 80B, 16B aligned

__global__ __launch_bounds__(256) void qkv_gemm_kernel(const float* __restrict__ X,
                                                       const float* __restrict__ W,
                                                       float* __restrict__ QKV) {
    __shared__ __align__(16) __bf16 Alds[2][128 * LDA];
    __shared__ __align__(16) __bf16 Blds[2][128 * LDA];   // B transposed: [n][k]

    const int tid  = threadIdx.x;
    const int lane = tid & 31;
    const int wave = tid >> 5;
    const int ln   = lane & 15;
    const int lh   = lane >> 4;
    const int wm   = wave & 3;    // 0..3 -> row 32*wm
    const int wn   = wave >> 2;   // 0..1 -> col 64*wn
    const int rb   = blockIdx.y * 128;
    const int cb   = blockIdx.x * 128;

    f32x8 acc[2][4];
    #pragma unroll
    for (int mt = 0; mt < 2; ++mt)
        #pragma unroll
        for (int nt = 0; nt < 4; ++nt)
            acc[mt][nt] = (f32x8){0.f,0.f,0.f,0.f,0.f,0.f,0.f,0.f};

    const int ar    = tid >> 1;          // 0..127  A row
    const int ah    = (tid & 1) * 16;    // k half within 32
    const int bk    = tid >> 3;          // 0..31   B k row
    const int bnoff = (tid & 7) * 16;    // n segment

    const float* Asrc = X + (size_t)(rb + ar) * 1024 + ah;      // + k0
    const float* Bsrc = W + (size_t)bk * 3072 + cb + bnoff;     // + k0*3072

    auto stage = [&](int buf, int k0) {
        // A tile [128 x 32] -> bf16 LDS row-major
        {
            const float* s = Asrc + k0;
            f32x4 r0 = *(const f32x4*)(s);
            f32x4 r1 = *(const f32x4*)(s + 4);
            f32x4 r2 = *(const f32x4*)(s + 8);
            f32x4 r3 = *(const f32x4*)(s + 12);
            f32x8 lo = __builtin_shufflevector(r0, r1, 0,1,2,3,4,5,6,7);
            f32x8 hi = __builtin_shufflevector(r2, r3, 0,1,2,3,4,5,6,7);
            bf16x8 p0 = __builtin_convertvector(lo, bf16x8);
            bf16x8 p1 = __builtin_convertvector(hi, bf16x8);
            __bf16* d = &Alds[buf][ar * LDA + ah];
            *(bf16x8*)d       = p0;
            *(bf16x8*)(d + 8) = p1;
        }
        // B tile W[k0..k0+31][cb..cb+127] transposed into Blds[n][k]
        {
            const float* s = Bsrc + (size_t)k0 * 3072;
            f32x4 r0 = *(const f32x4*)(s);
            f32x4 r1 = *(const f32x4*)(s + 4);
            f32x4 r2 = *(const f32x4*)(s + 8);
            f32x4 r3 = *(const f32x4*)(s + 12);
            bf16x4 c0 = __builtin_convertvector(r0, bf16x4);
            bf16x4 c1 = __builtin_convertvector(r1, bf16x4);
            bf16x4 c2 = __builtin_convertvector(r2, bf16x4);
            bf16x4 c3 = __builtin_convertvector(r3, bf16x4);
            __bf16* d = &Blds[buf][0];
            #pragma unroll
            for (int i = 0; i < 4; ++i) d[(bnoff +      i) * LDA + bk] = c0[i];
            #pragma unroll
            for (int i = 0; i < 4; ++i) d[(bnoff +  4 + i) * LDA + bk] = c1[i];
            #pragma unroll
            for (int i = 0; i < 4; ++i) d[(bnoff +  8 + i) * LDA + bk] = c2[i];
            #pragma unroll
            for (int i = 0; i < 4; ++i) d[(bnoff + 12 + i) * LDA + bk] = c3[i];
        }
    };

    stage(0, 0);
    __syncthreads();

    for (int k0 = 0; k0 < 1024; k0 += 32) {
        const int cur = (k0 >> 5) & 1;
        if (k0 + 32 < 1024) {
            stage(cur ^ 1, k0 + 32);
            if (k0 + 64 < 1024) {   // gfx1250 global_prefetch_b8, two steps ahead
                __builtin_prefetch(Asrc + k0 + 64, 0, 1);
                __builtin_prefetch(Bsrc + (size_t)(k0 + 64) * 3072, 0, 1);
            }
        }
        bf16x16 af0 = frag_lds(&Alds[cur][(wm * 32 +      ln) * LDA], lh * 8);
        bf16x16 af1 = frag_lds(&Alds[cur][(wm * 32 + 16 + ln) * LDA], lh * 8);
        #pragma unroll
        for (int nt = 0; nt < 4; ++nt) {
            bf16x16 bf = frag_lds(&Blds[cur][(wn * 64 + nt * 16 + ln) * LDA], lh * 8);
            acc[0][nt] = wmma_bf16(af0, bf, acc[0][nt]);
            acc[1][nt] = wmma_bf16(af1, bf, acc[1][nt]);
        }
        __syncthreads();
    }

    // epilogue: C layout -> lane N=ln, VGPR j -> row (mt*16 + lh*8 + j)
    #pragma unroll
    for (int mt = 0; mt < 2; ++mt) {
        #pragma unroll
        for (int nt = 0; nt < 4; ++nt) {
            const size_t row = (size_t)(rb + wm * 32 + mt * 16 + lh * 8);
            const int    col = cb + wn * 64 + nt * 16 + ln;
            float* o = QKV + row * 3072 + col;
            #pragma unroll
            for (int j = 0; j < 8; ++j)
                o[(size_t)j * 3072] = acc[mt][nt][j];
        }
    }
}

// ---------------------------------------------------------------------------
// Kernel 2: flash-style causal attention.
// Block = 32 query rows; 8 waves each own a 128-wide d-slice of O (in regs).
// Q staged ONCE per block in LDS (bf16). S = Q K^T split across waves
// (2x2 tiles x two 512-d halves), reduced in LDS. Online softmax; P bf16 in
// LDS; V tile transposed into LDS for PV WMMA.
// ---------------------------------------------------------------------------
__global__ __launch_bounds__(256) void flash_attn_kernel(const float* __restrict__ QKV,
                                                         float* __restrict__ Out) {
    __shared__ float Sp0[32][33];
    __shared__ float Sp1[32][33];
    __shared__ __align__(16) __bf16 Pb[32][32];       // P, row m x key
    __shared__ __align__(16) __bf16 Qlds[32 * 1024];  // [m][d] bf16, staged once
    __shared__ __align__(16) __bf16 Vt[32 * 1024];    // [d][key] transposed
    __shared__ __align__(16) float mrow[32];
    __shared__ __align__(16) float lrow[32];
    __shared__ __align__(16) float arow[32];

    const int tid  = threadIdx.x;
    const int lane = tid & 31;
    const int wave = tid >> 5;
    const int ln   = lane & 15;
    const int lh   = lane >> 4;
    const int qt   = blockIdx.x;          // 0..63 query tile
    const int bb   = blockIdx.y;          // 0..3 batch
    const int q0   = qt * 32;
    const int d0   = wave * 128;          // this wave's output d-slice

    const float* qbase = QKV + (size_t)bb * 2048 * 3072;        // q cols 0..1023
    const float* kbase = qbase + 1024;                          // k cols
    const float* vbase = qbase + 2048;                          // v cols

    f32x8 acc[2][8];
    #pragma unroll
    for (int mt = 0; mt < 2; ++mt)
        #pragma unroll
        for (int nt = 0; nt < 8; ++nt)
            acc[mt][nt] = (f32x8){0.f,0.f,0.f,0.f,0.f,0.f,0.f,0.f};

    if (tid < 32) { mrow[tid] = -INFINITY; lrow[tid] = 0.f; }

    // ---- stage Q tile [32 x 1024] f32 -> bf16 LDS (row-major), once per block
    #pragma unroll 4
    for (int it = 0; it < 32; ++it) {
        const int flat = tid + (it << 8);       // 0..8191 float4s
        const int row  = flat >> 8;
        const int dd   = (flat & 255) << 2;
        f32x4 qv = *(const f32x4*)(qbase + (size_t)(q0 + row) * 3072 + dd);
        *(bf16x4*)&Qlds[row * 1024 + dd] = __builtin_convertvector(qv, bf16x4);
    }
    __syncthreads();

    // S-tile assignment: waves 0-3 -> 4 tiles over d[0,512); waves 4-7 over d[512,1024)
    const int smt = (wave >> 1) & 1;
    const int snt = wave & 1;
    const int dlo = (wave >> 2) * 512;
    float* spart = (wave < 4) ? &Sp0[0][0] : &Sp1[0][0];
    const __bf16* qrow = &Qlds[(smt * 16 + ln) * 1024];

    for (int kt = 0; kt <= qt; ++kt) {
        const int k0 = kt * 32;

        // ---- Phase 1: partial S = Q K^T (this wave: one 16x16 tile, 512 of d)
        {
            f32x8 s = (f32x8){0.f,0.f,0.f,0.f,0.f,0.f,0.f,0.f};
            const float* kr = kbase + (size_t)(k0 + snt * 16 + ln) * 3072;
            #pragma unroll 4
            for (int ks = 0; ks < 16; ++ks) {
                const int dd = dlo + ks * 32;
                bf16x16 a  = frag_lds(qrow + dd, lh * 8);
                bf16x16 bk = frag_gbl_f32(kr + dd, lh * 8);
                s = wmma_bf16(a, bk, s);
            }
            #pragma unroll
            for (int j = 0; j < 8; ++j)
                spart[(smt * 16 + lh * 8 + j) * 33 + snt * 16 + ln] = s[j];
        }
        __syncthreads();

        // ---- Phase 2: combine halves, scale by 1/sqrt(1024), causal mask
        for (int idx = tid; idx < 1024; idx += 256) {
            const int r = idx >> 5, c = idx & 31;
            float v = (Sp0[r][c] + Sp1[r][c]) * 0.03125f;
            if (kt == qt && c > r) v = -INFINITY;
            Sp0[r][c] = v;
        }
        __syncthreads();

        // ---- Phase 3a: online softmax update (one thread per query row)
        if (tid < 32) {
            const int r = tid;
            const float mo = mrow[r];
            float rm = -INFINITY;
            #pragma unroll 8
            for (int c = 0; c < 32; ++c) rm = fmaxf(rm, Sp0[r][c]);
            const float mn = fmaxf(mo, rm);
            const float al = __expf(mo - mn);
            float ls = 0.f;
            #pragma unroll 8
            for (int c = 0; c < 32; ++c) {
                const float p = __expf(Sp0[r][c] - mn);
                Pb[r][c] = (__bf16)p;
                ls += p;
            }
            mrow[r] = mn;
            lrow[r] = lrow[r] * al + ls;
            arow[r] = al;
        }
        // ---- Phase 3b: stage V tile [32 keys x 1024 d] transposed -> Vt[d][key]
        #pragma unroll 4
        for (int it = 0; it < 32; ++it) {
            const int flat = tid + (it << 8);       // 0..8191 float4s
            const int key  = flat >> 8;
            const int dd   = (flat & 255) << 2;
            f32x4 vv = *(const f32x4*)(vbase + (size_t)(k0 + key) * 3072 + dd);
            bf16x4 cv = __builtin_convertvector(vv, bf16x4);
            Vt[(dd + 0) * 32 + key] = cv[0];
            Vt[(dd + 1) * 32 + key] = cv[1];
            Vt[(dd + 2) * 32 + key] = cv[2];
            Vt[(dd + 3) * 32 + key] = cv[3];
        }
        __syncthreads();

        // ---- Phase 4: rescale O by alpha, accumulate O += P x V (this d-slice)
        {
            #pragma unroll
            for (int mt = 0; mt < 2; ++mt) {
                const float* ap = &arow[mt * 16 + lh * 8];
                f32x4 a0 = *(const f32x4*)ap;
                f32x4 a1 = *(const f32x4*)(ap + 4);
                f32x8 am;
                am[0]=a0[0]; am[1]=a0[1]; am[2]=a0[2]; am[3]=a0[3];
                am[4]=a1[0]; am[5]=a1[1]; am[6]=a1[2]; am[7]=a1[3];
                #pragma unroll
                for (int nt = 0; nt < 8; ++nt) acc[mt][nt] *= am;
            }
            bf16x16 pf0 = frag_lds(&Pb[ln][0],      lh * 8);
            bf16x16 pf1 = frag_lds(&Pb[16 + ln][0], lh * 8);
            #pragma unroll
            for (int nt = 0; nt < 8; ++nt) {
                bf16x16 vf = frag_lds(&Vt[(size_t)(d0 + nt * 16 + ln) * 32], lh * 8);
                acc[0][nt] = wmma_bf16(pf0, vf, acc[0][nt]);
                acc[1][nt] = wmma_bf16(pf1, vf, acc[1][nt]);
            }
        }
        // No trailing barrier needed: next iteration's LDS writes are all
        // protected by the phase-internal barriers above.
    }

    // ---- Epilogue: O /= l, write out
    #pragma unroll
    for (int mt = 0; mt < 2; ++mt) {
        const float* lp = &lrow[mt * 16 + lh * 8];
        f32x4 l0 = *(const f32x4*)lp;
        f32x4 l1 = *(const f32x4*)(lp + 4);
        f32x8 iv;
        iv[0]=1.f/l0[0]; iv[1]=1.f/l0[1]; iv[2]=1.f/l0[2]; iv[3]=1.f/l0[3];
        iv[4]=1.f/l1[0]; iv[5]=1.f/l1[1]; iv[6]=1.f/l1[2]; iv[7]=1.f/l1[3];
        #pragma unroll
        for (int nt = 0; nt < 8; ++nt) {
            const int col = d0 + nt * 16 + ln;
            float* op = Out + ((size_t)bb * 2048 + q0 + mt * 16 + lh * 8) * 1024 + col;
            #pragma unroll
            for (int j = 0; j < 8; ++j)
                op[(size_t)j * 1024] = acc[mt][nt][j] * iv[j];
        }
    }
}

// ---------------------------------------------------------------------------
extern "C" void kernel_launch(void* const* d_in, const int* in_sizes, int n_in,
                              void* d_out, int out_size, void* d_ws, size_t ws_size,
                              hipStream_t stream) {
    const float* x = (const float*)d_in[0];   // [4,2048,1024]
    const float* W = (const float*)d_in[1];   // [1024,3072]
    float* out = (float*)d_out;               // [4,2048,1024]
    float* qkv = (float*)d_ws;                // [8192,3072] f32 = 100.7 MB scratch

    dim3 g1(3072 / 128, 8192 / 128);          // (24, 64)
    qkv_gemm_kernel<<<g1, 256, 0, stream>>>(x, W, qkv);

    dim3 g2(2048 / 32, 4);                    // (64, 4)
    flash_attn_kernel<<<g2, 256, 0, stream>>>(qkv, out);
}
